// Plasmodium_50027779064181
// MI455X (gfx1250) — matrix operand order repaired and checked by reading, not compiled, not run on hardware
//
#include <hip/hip_runtime.h>
#include <hip/hip_bf16.h>

// MI455X / gfx1250, wave32. All matrix math via v_wmma_f32_16x16x32_f16.
// Workspace requirement: 6 * 2048 * 2048 * 4 B = 96 MB in d_ws.

typedef __attribute__((ext_vector_type(16))) _Float16 v16h;
typedef __attribute__((ext_vector_type(8)))  float    v8f;

__device__ __forceinline__ v8f wmma_f16(v16h a, v16h b, v8f c) {
  return __builtin_amdgcn_wmma_f32_16x16x32_f16(false, a, false, b, (short)0, c,
                                                false, false);
}

// Packed f32x2 -> f16x2 store (dst must be 4-byte aligned).
__device__ __forceinline__ void store_h2(_Float16* dst, float lo, float hi) {
  union { _Float16 h[2]; unsigned u; } u_;
  u_.h[0] = (_Float16)lo;
  u_.h[1] = (_Float16)hi;
  *(unsigned*)dst = u_.u;
}

// A-fragment (16x32 f16): lane-half hs holds K = [hs*16, hs*16+16).
// p must point at rowBase + hs*16 (contiguous halfs).
__device__ __forceinline__ v16h frag_a_h(const _Float16* p) {
  v16h a;
#pragma unroll
  for (int i = 0; i < 16; ++i) a[i] = p[i];
  return a;
}

// B-fragment (32x16 f16): lane n = lane&15 holds column n, K = [hs*16, hs*16+16).
// p must point at colBase + hs*16 (contiguous halfs).
__device__ __forceinline__ v16h frag_b_h(const _Float16* p) {
  v16h b;
#pragma unroll
  for (int i = 0; i < 16; ++i) b[i] = p[i];
  return b;
}

// Same as frag_b_h but converting from f32 (streamed from global/L2).
__device__ __forceinline__ v16h frag_b_f(const float* p) {
  v16h b;
#pragma unroll
  for (int i = 0; i < 16; ++i) b[i] = (_Float16)p[i];
  return b;
}

// ---------------------------------------------------------------------------
// Generic WMMA GEMM: C = act(A @ B(^T) + bias)
// Block tile 128x128, 256 threads = 8 waves, wave = 32(M) x 64(N)
//   -> 8 accumulator tiles / wave -> 8 WMMAs per 32-wide K-chunk per wave.
// Requires M % 128 == 0, N % 128 == 0, K % 32 == 0 (true for all shapes here).
// TRANSB=0: B stored [K][N] row-major.  TRANSB=1: B stored [N][K] row-major.
// ACT=1: tanh epilogue.
// ---------------------------------------------------------------------------
template <int TRANSB, int ACT>
__global__ __launch_bounds__(256) void gemm_kernel(
    const float* __restrict__ A, int lda, long sA,
    const float* __restrict__ B, int ldb, long sB,
    const float* __restrict__ bias,
    float* __restrict__ C, int ldc, long sC,
    int M, int N, int K) {
  A += (size_t)blockIdx.z * (size_t)sA;
  B += (size_t)blockIdx.z * (size_t)sB;
  C += (size_t)blockIdx.z * (size_t)sC;
  const int n0 = blockIdx.x * 128, m0 = blockIdx.y * 128;
  const int t = threadIdx.x, wave = t >> 5, lane = t & 31;
  const int hs = lane >> 4, mr = lane & 15;
  const int wM = (wave >> 1) * 32;   // 4 wave rows of 32
  const int wN = (wave & 1) * 64;    // 2 wave cols of 64

  __shared__ _Float16 Al[128][34];   // [m][k], padded -> conflict-free frags
  __shared__ _Float16 Bl[128][34];   // [n][k] (column-major staging of B tile)

  v8f acc[2][4] = {};

  // Per-thread staging coordinates (loop-invariant).
  const int srow = t >> 1, sc0 = (t & 1) * 16;
  const float* pA = A + (size_t)(m0 + srow) * lda + sc0;
  _Float16* dA = &Al[srow][sc0];

  for (int kk = 0; kk < K; kk += 32) {
    __syncthreads();
    {  // stage A tile 128x32 (f32 -> f16, packed b32 LDS stores)
      const float* p = pA + kk;
      __builtin_prefetch(p + 32, 0, 3);  // global_prefetch_b8: next K-chunk
#pragma unroll
      for (int j = 0; j < 8; ++j)
        store_h2(dA + 2 * j, p[2 * j], p[2 * j + 1]);
    }
    if (TRANSB) {  // B tile rows are n, contiguous in k: same pattern as A
      const float* p = B + (size_t)(n0 + srow) * ldb + kk + sc0;
      __builtin_prefetch(p + 32, 0, 3);
      _Float16* d = &Bl[srow][sc0];
#pragma unroll
      for (int j = 0; j < 8; ++j)
        store_h2(d + 2 * j, p[2 * j], p[2 * j + 1]);
    } else {
      // B stored [K][N]: each thread owns an even k-pair x 4 columns so the
      // transposed store packs (k, k+1) into one b32 per column.
      const int kr0 = (t >> 5) * 2, nc = lane * 4;
#pragma unroll
      for (int k2 = 0; k2 < 2; ++k2) {
        const int kr = kr0 + k2 * 16;
        const float* p0 = B + (size_t)(kk + kr) * ldb + n0 + nc;
        const float* p1 = p0 + ldb;
        __builtin_prefetch(p0 + 32 * (size_t)ldb, 0, 3);
        _Float16* d = &Bl[nc][kr];
#pragma unroll
        for (int i = 0; i < 4; ++i)
          store_h2(d + i * 34, p0[i], p1[i]);
      }
    }
    __syncthreads();

    v16h a0 = frag_a_h(&Al[wM + mr][hs * 16]);
    v16h a1 = frag_a_h(&Al[wM + 16 + mr][hs * 16]);
#pragma unroll
    for (int j = 0; j < 4; ++j) {
      v16h bf = frag_b_h(&Bl[wN + j * 16 + mr][hs * 16]);
      acc[0][j] = wmma_f16(a0, bf, acc[0][j]);
      acc[1][j] = wmma_f16(a1, bf, acc[1][j]);
    }
  }

#pragma unroll
  for (int j = 0; j < 4; ++j) {
    const int col = n0 + wN + j * 16 + mr;
    const float bv = bias ? bias[col] : 0.0f;
#pragma unroll
    for (int r = 0; r < 8; ++r) {
      const int row = m0 + wM + r + hs * 8;  // C layout: M = r + hs*8
      float o0 = acc[0][j][r] + bv;
      float o1 = acc[1][j][r] + bv;
      if (ACT) { o0 = tanhf(o0); o1 = tanhf(o1); }
      C[(size_t)row * ldc + col] = o0;
      C[(size_t)(row + 16) * ldc + col] = o1;
    }
  }
}

// ---------------------------------------------------------------------------
// Per-segment L2 normalize: dst[s] = src[s] / sqrt(sum(src[s]^2) + 1e-10),
// segment = contiguous d floats (head-concat rows make (row,head) contiguous).
// ---------------------------------------------------------------------------
__global__ __launch_bounds__(128) void l2norm_kernel(
    const float* __restrict__ src, float* __restrict__ dst, int d) {
  __shared__ float red[128];
  const size_t base = (size_t)blockIdx.x * d;
  const int t = threadIdx.x;
  float ss = 0.0f;
  for (int j = t; j < d; j += 128) { const float x = src[base + j]; ss += x * x; }
  red[t] = ss;
  __syncthreads();
  for (int s = 64; s > 0; s >>= 1) {
    if (t < s) red[t] += red[t + s];
    __syncthreads();
  }
  const float rn = 1.0f / sqrtf(red[0] + 1e-10f);
  for (int j = t; j < d; j += 128) dst[base + j] = src[base + j] * rn;
}

// ---------------------------------------------------------------------------
// Row LayerNorm in place: a = g*(a-mu)/sqrt(var+1e-5) + b
// ---------------------------------------------------------------------------
__global__ __launch_bounds__(256) void layernorm_kernel(
    float* __restrict__ a, const float* __restrict__ g,
    const float* __restrict__ b, int Lw) {
  __shared__ float r1[256], r2[256];
  const size_t base = (size_t)blockIdx.x * Lw;
  const int t = threadIdx.x;
  float s = 0.0f, s2 = 0.0f;
  for (int j = t; j < Lw; j += 256) {
    const float x = a[base + j];
    s += x; s2 += x * x;
  }
  r1[t] = s; r2[t] = s2;
  __syncthreads();
  for (int st = 128; st > 0; st >>= 1) {
    if (t < st) { r1[t] += r1[t + st]; r2[t] += r2[t + st]; }
    __syncthreads();
  }
  const float mu = r1[0] / (float)Lw;
  const float var = r2[0] / (float)Lw - mu * mu;
  const float rstd = 1.0f / sqrtf(var + 1e-5f);
  for (int j = t; j < Lw; j += 256)
    a[base + j] = g[j] * (a[base + j] - mu) * rstd + b[j];
}

// ---------------------------------------------------------------------------
// Flash-style cosine attention. Block = (head p, 16 query rows), 8 waves.
// Waves 0-1 compute the 16x32 score tile (qn tile in LDS, kn from L2),
// 16 threads do the online softmax update, all 8 waves accumulate exp(P)@V
// with each wave owning a 64-wide d-chunk (4 f32 accum tiles per wave).
// a_out[row, p*512 + :] = sum_m exp(s-max) * v[m,:] / (sum exp + 1e-10)
// ---------------------------------------------------------------------------
__global__ __launch_bounds__(256) void attn_kernel(
    const float* __restrict__ qn, const float* __restrict__ kn,
    const float* __restrict__ v, float* __restrict__ a_out) {
  constexpr int Dh = 512, Lw = 2048, Nn = 2048;
  const int row0 = blockIdx.x * 16;
  const int hd = blockIdx.y * Dh;
  const int t = threadIdx.x, wave = t >> 5, lane = t & 31;
  const int hs = lane >> 4, mr = lane & 15;

  __shared__ _Float16 Ql[16 * 520];  // q tile [16][512], stride 520 (8-half pad)
  __shared__ _Float16 Vl[512 * 34];  // v block [dcol][key], stride 34
  __shared__ float    Sl[16 * 32];   // raw scores
  __shared__ _Float16 Pl[16 * 34];   // exp(P) tile, stride 34
  __shared__ float m_s[16], l_s[16], sc_s[16];

  {  // stage q tile (f32 -> f16, packed b32 stores)
    const int row = t >> 4, c0 = (t & 15) * 32;
    const float* p = qn + (size_t)(row0 + row) * Lw + hd + c0;
    _Float16* d = &Ql[row * 520 + c0];
#pragma unroll 8
    for (int j = 0; j < 16; ++j)
      store_h2(d + 2 * j, p[2 * j], p[2 * j + 1]);
  }
  if (t < 16) { m_s[t] = -1e30f; l_s[t] = 0.0f; }

  v8f acc[4] = {};
  const float sscale = 0.044194173824159216f;  // 1/sqrt(512) / TEMPERATURE

  // Loop-invariant staging coordinates for the V block.
  const int vkp = (t >> 4) * 2, vc0 = (t & 15) * 32;
  const float* pV = v + (size_t)vkp * Lw + hd + vc0;
  _Float16* dV = &Vl[vc0 * 34 + vkp];
  __syncthreads();

  for (int kb = 0; kb < Nn; kb += 32) {
    {  // stage v block 32x512 transposed; each thread owns an adjacent key
       // pair so the [dcol][key] scatter packs into b32 stores, and LDS
       // addresses are base + constant offsets (fold into ds_store imm).
      const float* p0 = pV + (size_t)kb * Lw;
      const float* p1 = p0 + Lw;
      __builtin_prefetch(p0 + 32 * (size_t)Lw, 0, 3);  // next key block
#pragma unroll 8
      for (int j = 0; j < 32; ++j)
        store_h2(dV + j * 34, p0[j], p1[j]);
    }
    if (wave < 2) {  // wave-uniform branch: EXEC stays all-1s for WMMA
      v8f s = {};
      const int key = kb + wave * 16 + mr;
      const float* kp = kn + (size_t)key * Lw + hd;
      __builtin_prefetch(kp + 32 * (size_t)Lw, 0, 3);  // next block's kn row
      for (int kc = 0; kc < Dh; kc += 32) {
        v16h aq = frag_a_h(&Ql[mr * 520 + kc + hs * 16]);
        v16h bk = frag_b_f(kp + kc + hs * 16);
        s = wmma_f16(aq, bk, s);
      }
#pragma unroll
      for (int r = 0; r < 8; ++r)
        Sl[(r + hs * 8) * 32 + wave * 16 + mr] = s[r] * sscale;
    }
    __syncthreads();
    if (t < 16) {  // online softmax update for row t
      const float mo = m_s[t];
      float mb = mo;
#pragma unroll
      for (int j = 0; j < 32; ++j) mb = fmaxf(mb, Sl[t * 32 + j]);
      const float sc = __expf(mo - mb);
      float sum = 0.0f;
      _Float16* dP = &Pl[t * 34];
#pragma unroll
      for (int j = 0; j < 16; ++j) {
        const float e0 = __expf(Sl[t * 32 + 2 * j] - mb);
        const float e1 = __expf(Sl[t * 32 + 2 * j + 1] - mb);
        store_h2(dP + 2 * j, e0, e1);
        sum += e0 + e1;
      }
      l_s[t] = l_s[t] * sc + sum;
      m_s[t] = mb;
      sc_s[t] = sc;
    }
    __syncthreads();
    {  // rescale accumulators, then acc += exp(P) @ V
      v16h pf = frag_a_h(&Pl[mr * 34 + hs * 16]);
#pragma unroll
      for (int r = 0; r < 8; ++r) {
        const float sc = sc_s[r + hs * 8];
#pragma unroll
        for (int t4 = 0; t4 < 4; ++t4) acc[t4][r] *= sc;
      }
#pragma unroll
      for (int t4 = 0; t4 < 4; ++t4) {
        const int dcol = wave * 64 + t4 * 16 + mr;
        v16h vf = frag_b_h(&Vl[dcol * 34 + hs * 16]);
        acc[t4] = wmma_f16(pf, vf, acc[t4]);
      }
    }
    __syncthreads();
  }

#pragma unroll
  for (int t4 = 0; t4 < 4; ++t4) {
#pragma unroll
    for (int r = 0; r < 8; ++r) {
      const int row = r + hs * 8;
      const int col = hd + wave * 64 + t4 * 16 + mr;
      a_out[(size_t)(row0 + row) * Lw + col] = acc[t4][r] / (l_s[row] + 1e-10f);
    }
  }
}

// ---------------------------------------------------------------------------
extern "C" void kernel_launch(void* const* d_in, const int* in_sizes, int n_in,
                              void* d_out, int out_size, void* d_ws,
                              size_t ws_size, hipStream_t stream) {
  constexpr int N = 2048, S = 1024, L = 2048, P = 4, Dh = 512;
  (void)in_sizes; (void)n_in; (void)out_size; (void)ws_size;

  const float* x   = (const float*)d_in[0];
  const float* We  = (const float*)d_in[1];
  const float* be  = (const float*)d_in[2];
  const float* Wq  = (const float*)d_in[3];
  const float* bq  = (const float*)d_in[4];
  const float* Wk  = (const float*)d_in[5];
  const float* bk  = (const float*)d_in[6];
  const float* Wv  = (const float*)d_in[7];
  const float* bv  = (const float*)d_in[8];
  const float* lng = (const float*)d_in[9];
  const float* lnb = (const float*)d_in[10];
  const float* Wd  = (const float*)d_in[11];
  const float* bd  = (const float*)d_in[12];

  float* out  = (float*)d_out;               // [N, S]
  float* corr = out + (size_t)N * S;         // [P, N, N]

  float* ws = (float*)d_ws;                  // 6 * N*L floats = 96 MB
  float* h  = ws;
  float* q  = h  + (size_t)N * L;            // becomes qn in place
  float* k  = q  + (size_t)N * L;            // becomes kn in place
  float* v  = k  + (size_t)N * L;
  float* vn = v  + (size_t)N * L;
  float* aa = vn + (size_t)N * L;            // attn concat; LN in place

  const dim3 blk(256);

  // h = tanh(x @ We + be)                [N,S]x[S,L]
  gemm_kernel<0, 1><<<dim3(L / 128, N / 128, 1), blk, 0, stream>>>(
      x, S, 0, We, L, 0, be, h, L, 0, N, L, S);
  // q = x @ Wq + bq
  gemm_kernel<0, 0><<<dim3(L / 128, N / 128, 1), blk, 0, stream>>>(
      x, S, 0, Wq, L, 0, bq, q, L, 0, N, L, S);
  // k = h @ Wk + bk                     [N,L]x[L,L]
  gemm_kernel<0, 0><<<dim3(L / 128, N / 128, 1), blk, 0, stream>>>(
      h, L, 0, Wk, L, 0, bk, k, L, 0, N, L, L);
  // v = h @ Wv + bv
  gemm_kernel<0, 0><<<dim3(L / 128, N / 128, 1), blk, 0, stream>>>(
      h, L, 0, Wv, L, 0, bv, v, L, 0, N, L, L);

  // per-(row,head) L2 normalization (segments are contiguous 512 floats)
  l2norm_kernel<<<dim3(N * P), dim3(128), 0, stream>>>(q, q, Dh);
  l2norm_kernel<<<dim3(N * P), dim3(128), 0, stream>>>(k, k, Dh);
  l2norm_kernel<<<dim3(N * P), dim3(128), 0, stream>>>(v, vn, Dh);

  // flash attention -> aa [N, L] (head-concat)
  attn_kernel<<<dim3(N / 16, P), blk, 0, stream>>>(q, k, v, aa);

  // LayerNorm in place
  layernorm_kernel<<<dim3(N), blk, 0, stream>>>(aa, lng, lnb, L);

  // out = tanh(aa @ Wd + bd)            [N,L]x[L,S]
  gemm_kernel<0, 1><<<dim3(S / 128, N / 128, 1), blk, 0, stream>>>(
      aa, L, 0, Wd, S, 0, bd, out, S, 0, N, S, L);

  // corr[p] = kn_p @ vn_p^T (batched over heads via grid.z, B transposed)
  gemm_kernel<1, 0><<<dim3(N / 128, N / 128, P), blk, 0, stream>>>(
      k, L, (long)Dh, vn, L, (long)Dh, nullptr, corr, N, (long)N * N, N, N, Dh);
}
